// JslBERT_33457795236754
// MI455X (gfx1250) — matrix operand order repaired and checked by reading, not compiled, not run on hardware
//
#include <hip/hip_runtime.h>
#include <math.h>
#include <stdint.h>

// ---------------------------------------------------------------------------
// Problem constants (BERT-like encoder)
// ---------------------------------------------------------------------------
#define BB 4
#define SS 512
#define LL 4
#define DD 768
#define HH 12
#define DHEAD 768               // per-head dim equals D in this reference
#define NQKV (HH * DHEAD)       // 9216
#define MROWS (BB * SS)         // 2048
#define EPS 1e-3f

// LDS tile geometry: 128 rows x 32 K-elements, padded to 40 elems (80 B) per
// row -> bank step of 20 dwords, conflict-free ds_load_b128 fragment reads.
#define KSTEP 32
#define LDS_ROW 40
#define TILE_ELEMS (128 * LDS_ROW)
#define TILE_BYTES (TILE_ELEMS * 2)       // 10240

typedef __attribute__((ext_vector_type(16))) __bf16 v16bf;
typedef __attribute__((ext_vector_type(8)))  __bf16 v8bf;
typedef __attribute__((ext_vector_type(8)))  float  v8f;

// ---------------------------------------------------------------------------
// CDNA5 async global->LDS copy (ASYNCcnt-tracked hardware copy path).
// Copies 16 B per lane. dst = LDS byte offset, src = 64-bit global address.
// ---------------------------------------------------------------------------
__device__ __forceinline__ void async_copy_b128(unsigned lds_off, const void* gsrc) {
    asm volatile("global_load_async_to_lds_b128 %0, %1, off"
                 :: "v"(lds_off), "v"((unsigned long long)(uintptr_t)gsrc)
                 : "memory");
}

// ---------------------------------------------------------------------------
// WMMA fragment loads from LDS (CDNA5 16x16x32 bf16 layouts, wave32)
//   A (16x32): lane l -> m=l&15, hi=l>>4; elems 0..7 @ K=hi*8, 8..15 @ 16+hi*8
//   B (as Bt rows): lane l -> n=l&15, hi=l>>4; elems 0..7 @ K=hi*8, 8..15 @ 16+hi*8
// ---------------------------------------------------------------------------
__device__ __forceinline__ v16bf lds_frag_a(const __bf16* p) {
    union { v16bf v; v8bf h[2]; } u;
    u.h[0] = *(const v8bf*)(p);
    u.h[1] = *(const v8bf*)(p + 16);
    return u.v;
}
__device__ __forceinline__ v16bf lds_frag_b(const __bf16* p) {
    union { v16bf v; v8bf h[2]; } u;
    u.h[0] = *(const v8bf*)(p);
    u.h[1] = *(const v8bf*)(p + 16);
    return u.v;
}

// ---------------------------------------------------------------------------
// Generic bf16 GEMM:  C[z] = A[z] * Bt[z]^T (+ bias)
//   A: M x K (row-major, lda); Bt: N x K (row-major, ldb) == B^T
//   C: M x N (row-major, ldc), fp32 and/or bf16 output
// Per-z operand offsets: off = (z/zh)*s1 + (z%zh)*s2 (covers batched heads).
// Block tile 128x128, 8 waves (4 across M x 2 across N), wave tile 32x64.
// Double-buffered async global->LDS staging, pipeline unrolled x2 so both
// buffers have static LDS offsets; 8 WMMAs per wave per 32-K step.
// Requires: M % 128 == 0, N % 128 == 0, K % 64 == 0 (all shapes here comply).
// ---------------------------------------------------------------------------
template<int OUTF, int OUTB, int HASBIAS>
__global__ __launch_bounds__(256)
void gemm_bf16_wmma(const __bf16* __restrict__ A,
                    const __bf16* __restrict__ Bt,
                    float* __restrict__ Cf,
                    __bf16* __restrict__ Cb,
                    const float* __restrict__ bias,
                    int M, int N, int K,
                    int lda, int ldb, int ldc,
                    int zh,
                    long sA1, long sA2,
                    long sB1, long sB2,
                    long sC1, long sC2)
{
    __shared__ __bf16 tiles[2][2][TILE_ELEMS];   // [buf][A/B] : 40 KB total

    const int z  = blockIdx.z;
    const long za = z / zh, zb = z % zh;
    A  += za * sA1 + zb * sA2;
    Bt += za * sB1 + zb * sB2;
    const long cOff = za * sC1 + zb * sC2;

    const int tid   = threadIdx.x;
    const int lane  = tid & 31;
    const int wave  = tid >> 5;           // 0..7
    const int waveM = wave & 3;           // 4 waves across M (32 rows each)
    const int waveN = wave >> 2;          // 2 waves across N (64 cols each)
    const int rowM0 = blockIdx.y * 128;
    const int colN0 = blockIdx.x * 128;
    const int mloc  = lane & 15;
    const int hi    = lane >> 4;

    const unsigned base0A = (unsigned)(uintptr_t)&tiles[0][0][0];
    const unsigned base0B = (unsigned)(uintptr_t)&tiles[0][1][0];
    const unsigned base1A = (unsigned)(uintptr_t)&tiles[1][0][0];
    const unsigned base1B = (unsigned)(uintptr_t)&tiles[1][1][0];

    // --- per-thread staging assignments (2 x 16B chunks per tile) ----------
    const int r0 = tid >> 2,        c0 = tid & 3;          // chunk 0
    const int r1 = (tid + 256) >> 2, c1 = (tid + 256) & 3; // chunk 1
    const __bf16* gA0 = A  + (long)(rowM0 + r0) * lda + c0 * 8;
    const __bf16* gA1 = A  + (long)(rowM0 + r1) * lda + c1 * 8;
    const __bf16* gB0 = Bt + (long)(colN0 + r0) * ldb + c0 * 8;
    const __bf16* gB1 = Bt + (long)(colN0 + r1) * ldb + c1 * 8;
    const unsigned oA0 = (unsigned)(r0 * (LDS_ROW * 2) + c0 * 16);
    const unsigned oA1 = (unsigned)(r1 * (LDS_ROW * 2) + c1 * 16);

    auto stage = [&](unsigned bA, unsigned bB, int k0) {
        async_copy_b128(bA + oA0, gA0 + k0);
        async_copy_b128(bA + oA1, gA1 + k0);
        async_copy_b128(bB + oA0, gB0 + k0);
        async_copy_b128(bB + oA1, gB1 + k0);
    };

    // --- per-wave fragment addresses (constant across K steps) -------------
    const __bf16* fA0_0 = &tiles[0][0][(waveM * 32 + mloc) * LDS_ROW + hi * 8];
    const __bf16* fA1_0 = fA0_0 + 16 * LDS_ROW;
    const __bf16* fB0_0 = &tiles[0][1][(waveN * 64 + mloc) * LDS_ROW + hi * 8];
    const __bf16* fA0_1 = fA0_0 + 2 * TILE_ELEMS;
    const __bf16* fA1_1 = fA1_0 + 2 * TILE_ELEMS;
    const __bf16* fB0_1 = fB0_0 + 2 * TILE_ELEMS;

    v8f acc[2][4] = {};

    auto compute = [&](const __bf16* fA0, const __bf16* fA1, const __bf16* fB0) {
        // Load all 6 fragments first so the ds_loads pipeline, then issue
        // 8 back-to-back WMMAs (XDL co-executes with the next stage's setup).
        const v16bf a0 = lds_frag_a(fA0);
        const v16bf a1 = lds_frag_a(fA1);
        const v16bf b0 = lds_frag_b(fB0);
        const v16bf b1 = lds_frag_b(fB0 + 16 * LDS_ROW);
        const v16bf b2 = lds_frag_b(fB0 + 32 * LDS_ROW);
        const v16bf b3 = lds_frag_b(fB0 + 48 * LDS_ROW);
        acc[0][0] = __builtin_amdgcn_wmma_f32_16x16x32_bf16(false, a0, false, b0, (short)0, acc[0][0], false, false);
        acc[1][0] = __builtin_amdgcn_wmma_f32_16x16x32_bf16(false, a1, false, b0, (short)0, acc[1][0], false, false);
        acc[0][1] = __builtin_amdgcn_wmma_f32_16x16x32_bf16(false, a0, false, b1, (short)0, acc[0][1], false, false);
        acc[1][1] = __builtin_amdgcn_wmma_f32_16x16x32_bf16(false, a1, false, b1, (short)0, acc[1][1], false, false);
        acc[0][2] = __builtin_amdgcn_wmma_f32_16x16x32_bf16(false, a0, false, b2, (short)0, acc[0][2], false, false);
        acc[1][2] = __builtin_amdgcn_wmma_f32_16x16x32_bf16(false, a1, false, b2, (short)0, acc[1][2], false, false);
        acc[0][3] = __builtin_amdgcn_wmma_f32_16x16x32_bf16(false, a0, false, b3, (short)0, acc[0][3], false, false);
        acc[1][3] = __builtin_amdgcn_wmma_f32_16x16x32_bf16(false, a1, false, b3, (short)0, acc[1][3], false, false);
    };

    // --- software pipeline, unrolled x2 (K % 64 == 0) ----------------------
    stage(base0A, base0B, 0);                       // prologue: tile 0 -> buf0

    for (int k0 = 0; k0 + 64 <= K; k0 += 64) {
        // half A: stage k0+32 -> buf1, compute k0 from buf0
        stage(base1A, base1B, k0 + 32);
        asm volatile("s_wait_asynccnt 4" ::: "memory");   // buf0 landed
        __syncthreads();
        compute(fA0_0, fA1_0, fB0_0);
        __syncthreads();                                  // buf0 free

        // half B: stage k0+64 -> buf0 (if any), compute k0+32 from buf1
        if (k0 + 64 < K) {
            stage(base0A, base0B, k0 + 64);
            asm volatile("s_wait_asynccnt 4" ::: "memory");
        } else {
            asm volatile("s_wait_asynccnt 0" ::: "memory");
        }
        __syncthreads();
        compute(fA0_1, fA1_1, fB0_1);
        __syncthreads();                                  // buf1 free
    }

    // Epilogue: C element (M = sm*16 + hi*8 + r, N = sn*16 + (lane&15))
    #pragma unroll
    for (int sm = 0; sm < 2; ++sm) {
        #pragma unroll
        for (int sn = 0; sn < 4; ++sn) {
            const int col = colN0 + waveN * 64 + sn * 16 + mloc;
            float bv = 0.0f;
            if (HASBIAS) bv = bias[col];
            #pragma unroll
            for (int r = 0; r < 8; ++r) {
                const int row = rowM0 + waveM * 32 + sm * 16 + hi * 8 + r;
                const float val = acc[sm][sn][r] + bv;
                const long idx = cOff + (long)row * ldc + col;
                if (OUTF) Cf[idx] = val;
                if (OUTB) Cb[idx] = (__bf16)val;
            }
        }
    }
}

// ---------------------------------------------------------------------------
// Transpose (+convert to bf16): in[z] is rowsK x colsN (ld_in, type TI),
// out[z] is colsN x rowsK (ld = rowsK). 32x32 LDS tile, block (32,8).
// ---------------------------------------------------------------------------
template<typename TI>
__global__ void transpose_to_bf16(const TI* __restrict__ in,
                                  __bf16* __restrict__ out,
                                  int ld_in, int rowsK, int colsN,
                                  int zh, long sI1, long sI2, long sOut)
{
    __shared__ float tile[32][33];
    const int z = blockIdx.z;
    const TI* ip = in + (long)(z / zh) * sI1 + (long)(z % zh) * sI2;
    __bf16* op = out + (long)z * sOut;
    const int tx = threadIdx.x, ty = threadIdx.y;
    const int r0 = blockIdx.y * 32, c0 = blockIdx.x * 32;
    #pragma unroll
    for (int i = 0; i < 4; ++i)
        tile[ty + i * 8][tx] = (float)ip[(long)(r0 + ty + i * 8) * ld_in + c0 + tx];
    __syncthreads();
    #pragma unroll
    for (int i = 0; i < 4; ++i)
        op[(long)(c0 + ty + i * 8) * rowsK + r0 + tx] = (__bf16)tile[tx][ty + i * 8];
}

// ---------------------------------------------------------------------------
// Embedding: res = tok_w[tok]+tok_b + pos_w[pos]+pos_b + seg_w[seg]+seg_b
// ---------------------------------------------------------------------------
__global__ void embed_kernel(const int* __restrict__ X,
                             const float* __restrict__ tok_w, const float* __restrict__ tok_b,
                             const float* __restrict__ pos_w, const float* __restrict__ pos_b,
                             const float* __restrict__ seg_w, const float* __restrict__ seg_b,
                             float* __restrict__ resf, __bf16* __restrict__ resb)
{
    const int row = blockIdx.x;                 // b*S + s
    const int b = row / SS, s = row % SS;
    const int tok = X[(long)b * 3 * SS + 0 * SS + s];
    const int pos = X[(long)b * 3 * SS + 1 * SS + s];
    const int seg = X[(long)b * 3 * SS + 2 * SS + s];
    for (int d = threadIdx.x; d < DD; d += 256) {
        float v = tok_w[(long)tok * DD + d] + tok_b[d]
                + pos_w[(long)pos * DD + d] + pos_b[d]
                + seg_w[(long)seg * DD + d] + seg_b[d];
        resf[(long)row * DD + d] = v;
        resb[(long)row * DD + d] = (__bf16)v;
    }
}

// ---------------------------------------------------------------------------
// Masked softmax over key axis (512) with 1/sqrt(D) scale. bf16 output.
// ---------------------------------------------------------------------------
__global__ void softmax_kernel(const float* __restrict__ scores,
                               const int* __restrict__ X,
                               __bf16* __restrict__ attn, float scale)
{
    __shared__ float red[256];
    const int row = blockIdx.x;                 // b*H*S + h*S + t
    const int b = row / (HH * SS);
    const int* tokrow = X + (long)b * 3 * SS;   // tok ids
    const long base = (long)row * SS;
    const int t = threadIdx.x;

    float x0 = scores[base + t] * scale;
    float x1 = scores[base + t + 256] * scale;
    if (tokrow[t] == 0)       x0 = -1.0e9f;
    if (tokrow[t + 256] == 0) x1 = -1.0e9f;

    red[t] = fmaxf(x0, x1);
    __syncthreads();
    for (int off = 128; off > 0; off >>= 1) {
        if (t < off) red[t] = fmaxf(red[t], red[t + off]);
        __syncthreads();
    }
    const float m = red[0];
    __syncthreads();

    const float e0 = __expf(x0 - m), e1 = __expf(x1 - m);
    red[t] = e0 + e1;
    __syncthreads();
    for (int off = 128; off > 0; off >>= 1) {
        if (t < off) red[t] += red[t + off];
        __syncthreads();
    }
    const float inv = 1.0f / red[0];
    attn[base + t]       = (__bf16)(e0 * inv);
    attn[base + t + 256] = (__bf16)(e1 * inv);
}

// ---------------------------------------------------------------------------
// LayerNorm over last dim (768): outf (fp32) + outb (bf16).
// ---------------------------------------------------------------------------
__global__ void layernorm_kernel(const float* __restrict__ in,
                                 const float* __restrict__ g,
                                 const float* __restrict__ bta,
                                 float* __restrict__ outf,
                                 __bf16* __restrict__ outb)
{
    __shared__ float rsum[256], rsq[256];
    const int row = blockIdx.x;
    const long base = (long)row * DD;
    const int t = threadIdx.x;
    float v[3], s = 0.0f, sq = 0.0f;
    #pragma unroll
    for (int i = 0; i < 3; ++i) {
        v[i] = in[base + t + i * 256];
        s += v[i]; sq += v[i] * v[i];
    }
    rsum[t] = s; rsq[t] = sq;
    __syncthreads();
    for (int off = 128; off > 0; off >>= 1) {
        if (t < off) { rsum[t] += rsum[t + off]; rsq[t] += rsq[t + off]; }
        __syncthreads();
    }
    const float mean = rsum[0] * (1.0f / DD);
    const float var  = rsq[0] * (1.0f / DD) - mean * mean;
    const float rs   = rsqrtf(var + EPS);
    #pragma unroll
    for (int i = 0; i < 3; ++i) {
        const int d = t + i * 256;
        const float o = (v[i] - mean) * rs * g[d] + bta[d];
        outf[base + d] = o;
        outb[base + d] = (__bf16)o;
    }
}

// ---------------------------------------------------------------------------
// Host orchestration
// ---------------------------------------------------------------------------
extern "C" void kernel_launch(void* const* d_in, const int* in_sizes, int n_in,
                              void* d_out, int out_size, void* d_ws, size_t ws_size,
                              hipStream_t stream) {
    (void)in_sizes; (void)n_in; (void)out_size; (void)ws_size;

    const int*   X     = (const int*)  d_in[0];
    const float* tok_w = (const float*)d_in[1];
    const float* tok_b = (const float*)d_in[2];
    const float* pos_w = (const float*)d_in[3];
    const float* pos_b = (const float*)d_in[4];
    const float* seg_w = (const float*)d_in[5];
    const float* seg_b = (const float*)d_in[6];
    const float* Wq    = (const float*)d_in[7];
    const float* bq    = (const float*)d_in[8];
    const float* Wk    = (const float*)d_in[9];
    const float* bk    = (const float*)d_in[10];
    const float* Wv    = (const float*)d_in[11];
    const float* bv    = (const float*)d_in[12];
    const float* Wo    = (const float*)d_in[13];
    const float* bo    = (const float*)d_in[14];
    const float* ln1_g = (const float*)d_in[15];
    const float* ln1_b = (const float*)d_in[16];
    const float* ffp_w = (const float*)d_in[17];
    const float* ffp_b = (const float*)d_in[18];
    const float* ln2_g = (const float*)d_in[19];
    const float* ln2_b = (const float*)d_in[20];

    float* resf = (float*)d_out;                     // residual lives in d_out

    // ---- carve workspace -------------------------------------------------
    char* w = (char*)d_ws;
    size_t off = 0;
    auto carve = [&](size_t bytes) -> void* {
        void* p = w + off;
        off = (off + bytes + 255) & ~(size_t)255;
        return p;
    };
    __bf16* resb   = (__bf16*)carve((size_t)MROWS * DD * 2);
    __bf16* qb     = (__bf16*)carve((size_t)MROWS * NQKV * 2);
    __bf16* kb     = (__bf16*)carve((size_t)MROWS * NQKV * 2);
    __bf16* vb     = (__bf16*)carve((size_t)MROWS * NQKV * 2);   // later reused as ctx
    __bf16* vt     = (__bf16*)carve((size_t)BB * HH * DHEAD * SS * 2);
    float*  scores = (float*) carve((size_t)BB * HH * SS * SS * 4);
    __bf16* attn   = (__bf16*)carve((size_t)BB * HH * SS * SS * 2);
    __bf16* WqT    = (__bf16*)carve((size_t)NQKV * DD * 2);
    __bf16* WkT    = (__bf16*)carve((size_t)NQKV * DD * 2);
    __bf16* WvT    = (__bf16*)carve((size_t)NQKV * DD * 2);
    __bf16* WoT    = (__bf16*)carve((size_t)DD * NQKV * 2);
    __bf16* ffpT   = (__bf16*)carve((size_t)DD * DD * 2);
    float*  aout   = (float*) carve((size_t)MROWS * DD * 4);
    float*  h1f    = (float*) carve((size_t)MROWS * DD * 4);
    __bf16* h1b    = (__bf16*)carve((size_t)MROWS * DD * 2);
    float*  ffout  = (float*) carve((size_t)MROWS * DD * 4);

    const dim3 blk256(256);
    const dim3 tblk(32, 8);
    const float scale = 1.0f / sqrtf((float)DD);

    // ---- embedding -------------------------------------------------------
    embed_kernel<<<dim3(MROWS), blk256, 0, stream>>>(
        X, tok_w, tok_b, pos_w, pos_b, seg_w, seg_b, resf, resb);

    for (int l = 0; l < LL; ++l) {
        const long wqkv_off = (long)l * DD * NQKV;
        const long wo_off   = (long)l * NQKV * DD;
        const long ff_off   = (long)l * DD * DD;

        // ---- per-layer weight transposes (fp32 -> bf16, K-contiguous Bt) --
        transpose_to_bf16<float><<<dim3(NQKV / 32, DD / 32, 1), tblk, 0, stream>>>(
            Wq + wqkv_off, WqT, NQKV, DD, NQKV, 1, 0, 0, 0);
        transpose_to_bf16<float><<<dim3(NQKV / 32, DD / 32, 1), tblk, 0, stream>>>(
            Wk + wqkv_off, WkT, NQKV, DD, NQKV, 1, 0, 0, 0);
        transpose_to_bf16<float><<<dim3(NQKV / 32, DD / 32, 1), tblk, 0, stream>>>(
            Wv + wqkv_off, WvT, NQKV, DD, NQKV, 1, 0, 0, 0);
        transpose_to_bf16<float><<<dim3(DD / 32, NQKV / 32, 1), tblk, 0, stream>>>(
            Wo + wo_off, WoT, DD, NQKV, DD, 1, 0, 0, 0);
        transpose_to_bf16<float><<<dim3(DD / 32, DD / 32, 1), tblk, 0, stream>>>(
            ffp_w + ff_off, ffpT, DD, DD, DD, 1, 0, 0, 0);

        // ---- Q/K/V projections: (2048x768)x(768x9216), bf16 out + bias ----
        gemm_bf16_wmma<0, 1, 1><<<dim3(NQKV / 128, MROWS / 128, 1), blk256, 0, stream>>>(
            resb, WqT, nullptr, qb, bq + (long)l * NQKV,
            MROWS, NQKV, DD, DD, DD, NQKV, 1, 0, 0, 0, 0, 0, 0);
        gemm_bf16_wmma<0, 1, 1><<<dim3(NQKV / 128, MROWS / 128, 1), blk256, 0, stream>>>(
            resb, WkT, nullptr, kb, bk + (long)l * NQKV,
            MROWS, NQKV, DD, DD, DD, NQKV, 1, 0, 0, 0, 0, 0, 0);
        gemm_bf16_wmma<0, 1, 1><<<dim3(NQKV / 128, MROWS / 128, 1), blk256, 0, stream>>>(
            resb, WvT, nullptr, vb, bv + (long)l * NQKV,
            MROWS, NQKV, DD, DD, DD, NQKV, 1, 0, 0, 0, 0, 0, 0);

        // ---- V^T per (b,h): (512 x 768) -> (768 x 512) --------------------
        transpose_to_bf16<__bf16><<<dim3(DHEAD / 32, SS / 32, BB * HH), tblk, 0, stream>>>(
            vb, vt, NQKV, SS, DHEAD,
            HH, (long)SS * NQKV, (long)DHEAD, (long)DHEAD * SS);

        // ---- scores = Q * K^T per (b,h): z = b*H + h ----------------------
        gemm_bf16_wmma<1, 0, 0><<<dim3(SS / 128, SS / 128, BB * HH), blk256, 0, stream>>>(
            qb, kb, scores, nullptr, nullptr,
            SS, SS, DHEAD, NQKV, NQKV, SS,
            HH,
            (long)SS * NQKV, (long)DHEAD,             // A offsets (b, h)
            (long)SS * NQKV, (long)DHEAD,             // Bt offsets
            (long)HH * SS * SS, (long)SS * SS);       // C: linear z*S*S

        // ---- softmax(scale, mask) -> bf16 attn ----------------------------
        softmax_kernel<<<dim3(BB * HH * SS), blk256, 0, stream>>>(scores, X, attn, scale);

        // ---- ctx = attn * V per (b,h); write into vb as (b,s)x(h,kk) ------
        gemm_bf16_wmma<0, 1, 0><<<dim3(DHEAD / 128, SS / 128, BB * HH), blk256, 0, stream>>>(
            attn, vt, nullptr, vb, nullptr,
            SS, DHEAD, SS, SS, SS, NQKV,
            HH,
            (long)HH * SS * SS, (long)SS * SS,        // A: linear z*S*S
            (long)HH * DHEAD * SS, (long)DHEAD * SS,  // Bt: linear z*Dh*S
            (long)SS * NQKV, (long)DHEAD);            // C: b*S*9216 + h*768

        // ---- output projection: (2048x9216)x(9216x768) + bo, fp32 ---------
        gemm_bf16_wmma<1, 0, 1><<<dim3(DD / 128, MROWS / 128, 1), blk256, 0, stream>>>(
            vb, WoT, aout, nullptr, bo + (long)l * DD,
            MROWS, DD, NQKV, NQKV, NQKV, DD, 1, 0, 0, 0, 0, 0, 0);

        // ---- LN1 ----------------------------------------------------------
        layernorm_kernel<<<dim3(MROWS), blk256, 0, stream>>>(
            aout, ln1_g + (long)l * DD, ln1_b + (long)l * DD, h1f, h1b);

        // ---- FF: (2048x768)x(768x768) + ffp_b, fp32 -----------------------
        gemm_bf16_wmma<1, 0, 1><<<dim3(DD / 128, MROWS / 128, 1), blk256, 0, stream>>>(
            h1b, ffpT, ffout, nullptr, ffp_b + (long)l * DD,
            MROWS, DD, DD, DD, DD, DD, 1, 0, 0, 0, 0, 0, 0);

        // ---- LN2 -> residual (fp32 in d_out, bf16 for next layer) ---------
        layernorm_kernel<<<dim3(MROWS), blk256, 0, stream>>>(
            ffout, ln2_g + (long)l * DD, ln2_b + (long)l * DD, resf, resb);
    }
}
